// Shared_LoraLinear_15015205667343
// MI455X (gfx1250) — matrix operand order repaired and checked by reading, not compiled, not run on hardware
//
#include <hip/hip_runtime.h>
#include <hip/hip_bf16.h>

// Problem constants (from reference): B=4, T=4096, D_IN=D_OUT=2048, R=64, scaling=2
#define MTOT   16384
#define KDIM   2048
#define NDIM   2048
#define RANK   64
#define SCALE  2.0f

typedef __attribute__((ext_vector_type(16))) __bf16 v16bf;
typedef __attribute__((ext_vector_type(8)))  __bf16 v8bf;
typedef __attribute__((ext_vector_type(2)))  __bf16 v2bf;
typedef __attribute__((ext_vector_type(2)))  float  v2f;
typedef __attribute__((ext_vector_type(8)))  float  v8f;

typedef __attribute__((ext_vector_type(4))) unsigned int u32x4;
typedef __attribute__((ext_vector_type(8))) int          i32x8;
typedef __attribute__((ext_vector_type(4))) int          i32x4;

#if __has_builtin(__builtin_amdgcn_tensor_load_to_lds)
#define USE_TDM 1
#else
#define USE_TDM 0
#endif

// ---------------------------------------------------------------------------
// Pass 1: fold LoRA into the weight:  W_eff[n,k] = W[n,k] + 2 * sum_r B[n,r]*A[r,k]
// (x A^T) B^T == x (B A)^T, so the whole op collapses to ONE GEMM vs W_eff.
// grid: (KDIM/256, NDIM), block: 256
// ---------------------------------------------------------------------------
__global__ __launch_bounds__(256)
void build_weff_kernel(const float* __restrict__ W,
                       const float* __restrict__ A,
                       const float* __restrict__ Bm,
                       __bf16* __restrict__ Weff) {
    const int o = blockIdx.y;
    const int k = blockIdx.x * 256 + threadIdx.x;

    __shared__ float bsh[RANK];
    if (threadIdx.x < RANK)
        bsh[threadIdx.x] = Bm[o * RANK + threadIdx.x] * SCALE;
    __syncthreads();

    float acc = W[(size_t)o * KDIM + k];
#pragma unroll 8
    for (int r = 0; r < RANK; ++r)
        acc = fmaf(bsh[r], A[(size_t)r * KDIM + k], acc);

    Weff[(size_t)o * KDIM + k] = (__bf16)acc;
}

// ---------------------------------------------------------------------------
// Pass 2: Out[M,N] = X[M,K] @ Weff[N,K]^T   (bf16 WMMA, f32 accumulate)
// 128x128 block tile, 8 waves (4Mx2N), wave tile 32x64 = 2x4 16x16 f32 accs.
// K tiles of 64 via double-buffered LDS, one barrier per K-iter.
//  - x tile: global b128 loads, v_cvt_pk_bf16_f32 in flight, ds_store.
//  - W tile: Tensor Data Mover descriptor (tensor_load_to_lds) issued by
//    wave 0; TDM padding (4 dwords per 32-dword row) produces the same
//    conflict-free 144B LDS pitch; synced with s_wait_tensorcnt + barrier.
// grid: (NDIM/128, MTOT/128), block: 256
// ---------------------------------------------------------------------------
#define LDS_STRIDE 72   // 64 + 8 pad elements; 144B pitch (16B aligned, conflict-free)

__device__ __forceinline__ uint2 cvt4_bf16(const float4 v) {
    v2f ab = {v.x, v.y};
    v2f cd = {v.z, v.w};
    v2bf lo = __builtin_convertvector(ab, v2bf);   // -> v_cvt_pk_bf16_f32
    v2bf hi = __builtin_convertvector(cd, v2bf);
    uint2 r;
    r.x = __builtin_bit_cast(unsigned int, lo);
    r.y = __builtin_bit_cast(unsigned int, hi);
    return r;
}

__global__ __launch_bounds__(256, 1)
void lora_gemm_kernel(const float* __restrict__ X,
                      const __bf16* __restrict__ Weff,
                      float* __restrict__ Out) {
    // double-buffered tiles: 2 * 18KB * 2 = 72 KB (WGP has 320 KB)
    __shared__ __attribute__((aligned(16))) __bf16 xs [2][128 * LDS_STRIDE];
    __shared__ __attribute__((aligned(16))) __bf16 wls[2][128 * LDS_STRIDE];

    const int t    = threadIdx.x;
    const int lane = t & 31;
    const int wid  = t >> 5;
    const int wm   = wid >> 1;          // 0..3 : wave M position
    const int wn   = wid & 1;           // 0..1 : wave N position
    const int m_blk = blockIdx.y * 128;
    const int n_blk = blockIdx.x * 128;

    const float4* Xv = reinterpret_cast<const float4*>(X);     // 512 float4 per row

    v8f acc[2][4];
#pragma unroll
    for (int i = 0; i < 2; ++i)
#pragma unroll
        for (int j = 0; j < 4; ++j)
            acc[i][j] = (v8f)(0.0f);

    uint2 xreg[8];    // 128x64 x-tile, bf16-converted at load (8 x 8B / thread)

    const int xrow = t >> 4, xcol = t & 15;   // x: 16 float4 per row

    auto load_x = [&](int kt) {
#pragma unroll
        for (int i = 0; i < 8; ++i)
            xreg[i] = cvt4_bf16(
                Xv[(size_t)(m_blk + xrow + i * 16) * 512 + kt * 16 + xcol]);
    };
    auto store_x = [&](int buf) {
#pragma unroll
        for (int i = 0; i < 8; ++i)
            *reinterpret_cast<uint2*>(&xs[buf][(xrow + i * 16) * LDS_STRIDE + xcol * 4]) = xreg[i];
    };

#if USE_TDM
    // One TDM descriptor moves the whole 128x64 bf16 W tile (16 KB) to LDS.
    auto issue_tdm_w = [&](int kt, int buf) {
        const unsigned lds_base =
            (unsigned)(size_t)(&wls[buf][0]);                  // flat addr low 32 = LDS byte addr
        const unsigned long long ga =
            (unsigned long long)(size_t)(Weff + (size_t)n_blk * KDIM + (size_t)kt * 64);

        u32x4 g0 = {};
        g0.x = 1u;                                             // count=1, user descriptor
        g0.y = lds_base;                                       // lds_addr [63:32]
        g0.z = (unsigned)(ga & 0xFFFFFFFFu);                   // global_addr [95:64]
        g0.w = (unsigned)((ga >> 32) & 0x01FFFFFFu)            // global_addr [120:96]
             | (2u << 30);                                     // type=2 ("image")

        i32x8 g1 = {};
        g1[0] = (1 << 16)                                      // data_size: 2 bytes
              | (1 << 20)                                      // pad_enable
              | (4 << 22)                                      // pad_interval: 32 dwords (=row)
              | (3 << 25);                                     // pad_amount: 4 dwords (=16B)
        g1[1] = (int)((KDIM & 0xFFFF) << 16);                  // tensor_dim0[15:0]
        g1[2] = (int)(((KDIM >> 16) & 0xFFFF)                  // tensor_dim0[31:16]
              | ((NDIM & 0xFFFF) << 16));                      // tensor_dim1[15:0]
        g1[3] = (int)(((NDIM >> 16) & 0xFFFF)                  // tensor_dim1[31:16]
              | (64 << 16));                                   // tile_dim0 = 64 (K)
        g1[4] = 128;                                           // tile_dim1 = 128 rows
        g1[5] = KDIM;                                          // tensor_dim0_stride[31:0]
        g1[6] = 0;
        g1[7] = 0;

        i32x4 gz4 = {};                                        // groups 2/3 unused (2D)
        i32x8 gz8 = {};                                        // extra descriptor words unused
        __builtin_amdgcn_tensor_load_to_lds(g0, g1, gz4, gz4, gz8, 0);
    };
#else
    const uint4* Wv = reinterpret_cast<const uint4*>(Weff);    // 256 uint4 per row
    uint4 wreg[4];
    const int wrow = t >> 3, wcol = t & 7;
    auto load_w = [&](int kt) {
#pragma unroll
        for (int i = 0; i < 4; ++i)
            wreg[i] = Wv[(size_t)(n_blk + wrow + i * 32) * 256 + kt * 8 + wcol];
    };
    auto store_w = [&](int buf) {
#pragma unroll
        for (int i = 0; i < 4; ++i)
            *reinterpret_cast<uint4*>(&wls[buf][(wrow + i * 32) * LDS_STRIDE + wcol * 8]) = wreg[i];
    };
#endif

    // ---- prologue: tile 0 ----
#if USE_TDM
    if (wid == 0) issue_tdm_w(0, 0);
    load_x(0);
    store_x(0);
    __builtin_amdgcn_s_wait_tensorcnt(0);
#else
    load_x(0);
    load_w(0);
    store_x(0);
    store_w(0);
#endif
    __syncthreads();

    const int KT = KDIM / 64;   // 32 K-tiles
    for (int kt = 0; kt < KT; ++kt) {
        const int  cur  = kt & 1;
        const bool more = (kt + 1) < KT;
#if USE_TDM
        if (more && wid == 0) issue_tdm_w(kt + 1, cur ^ 1);    // async DMA into other buffer
#endif
        if (more) load_x(kt + 1);                              // register prefetch of next x
        if (kt + 2 < KT)                                       // L2 prefetch two ahead
            __builtin_prefetch(&Xv[(size_t)(m_blk + xrow) * 512 + (kt + 2) * 16 + xcol], 0, 0);
#if !USE_TDM
        if (more) load_w(kt + 1);
#endif

        // Two K=32 WMMA steps per K-tile of 64
#pragma unroll
        for (int ks = 0; ks < 2; ++ks) {
            // A fragments (16x32 bf16): lanes 0-15 hold K=k0..k0+7 & k0+16..k0+23,
            // lanes 16-31 shifted by +8 -> two contiguous 16B LDS reads.
            v16bf afr[2];
#pragma unroll
            for (int mi = 0; mi < 2; ++mi) {
                const __bf16* p = &xs[cur][(wm * 32 + mi * 16 + (lane & 15)) * LDS_STRIDE
                                           + ks * 32 + (lane >> 4) * 8];
                v8bf lo = *reinterpret_cast<const v8bf*>(p);
                v8bf hi = *reinterpret_cast<const v8bf*>(p + 16);
                afr[mi] = __builtin_shufflevector(lo, hi,
                          0,1,2,3,4,5,6,7,8,9,10,11,12,13,14,15);
            }
            // One B fragment live at a time: lane holds N=lane%16, K=k0..k0+7 & k0+16..k0+23
            // (lanes>=16 shifted by +8) -> two contiguous 16B LDS reads.
#pragma unroll
            for (int ni = 0; ni < 4; ++ni) {
                const __bf16* p = &wls[cur][(wn * 64 + ni * 16 + (lane & 15)) * LDS_STRIDE
                                            + ks * 32 + (lane >> 4) * 8];
                v8bf lo = *reinterpret_cast<const v8bf*>(p);
                v8bf hi = *reinterpret_cast<const v8bf*>(p + 16);
                v16bf bfr = __builtin_shufflevector(lo, hi,
                            0,1,2,3,4,5,6,7,8,9,10,11,12,13,14,15);
                acc[0][ni] = __builtin_amdgcn_wmma_f32_16x16x32_bf16(
                    false, afr[0], false, bfr, (short)0, acc[0][ni], false, false);
                acc[1][ni] = __builtin_amdgcn_wmma_f32_16x16x32_bf16(
                    false, afr[1], false, bfr, (short)0, acc[1][ni], false, false);
            }
        }

        // write next tile into the OTHER buffer -> single barrier per iter
        if (more) store_x(cur ^ 1);
#if USE_TDM
        if (more) __builtin_amdgcn_s_wait_tensorcnt(0);        // TDM done before barrier
#else
        if (more) store_w(cur ^ 1);
#endif
        __syncthreads();
    }

    // Epilogue: C/D layout — lane%16 = N column, VGPR v + 8*(lane>=16) = M row.
#pragma unroll
    for (int mi = 0; mi < 2; ++mi) {
#pragma unroll
        for (int ni = 0; ni < 4; ++ni) {
            const int col   = n_blk + wn * 64 + ni * 16 + (lane & 15);
            const int rbase = m_blk + wm * 32 + mi * 16 + ((lane >> 4) << 3);
#pragma unroll
            for (int v = 0; v < 8; ++v)
                Out[(size_t)(rbase + v) * NDIM + col] = acc[mi][ni][v];
        }
    }
}

// ---------------------------------------------------------------------------
extern "C" void kernel_launch(void* const* d_in, const int* in_sizes, int n_in,
                              void* d_out, int out_size, void* d_ws, size_t ws_size,
                              hipStream_t stream) {
    (void)in_sizes; (void)n_in; (void)out_size; (void)ws_size;
    const float* x  = (const float*)d_in[0];   // [4,4096,2048]
    const float* w  = (const float*)d_in[1];   // [2048,2048]
    const float* lA = (const float*)d_in[2];   // [64,2048]
    const float* lB = (const float*)d_in[3];   // [2048,64]
    float* out = (float*)d_out;                // [4,4096,2048]

    __bf16* weff = (__bf16*)d_ws;              // 2048*2048*2 = 8 MB scratch

    dim3 g1(KDIM / 256, NDIM);
    build_weff_kernel<<<g1, 256, 0, stream>>>(w, lA, lB, weff);

    dim3 g2(NDIM / 128, MTOT / 128);           // N fastest -> x tiles stay L2-resident
    lora_gemm_kernel<<<g2, 256, 0, stream>>>(x, weff, out);
}
